// HierarchicalSlotModel_38448547234696
// MI455X (gfx1250) — matrix operand order repaired and checked by reading, not compiled, not run on hardware
//
#include <hip/hip_runtime.h>
#include <hip/hip_bf16.h>
#include <math.h>

// ---------------------------------------------------------------------------
// MI455X (gfx1250) implementation.
//
// Key algorithmic transforms (exact up to FP reassociation):
//  1) conv1 input has only 10 distinct values (color embeddings) ->
//     precompute T[9][11][128] = emb[c] @ W1[ky][kx] (color 10 == zero pad).
//     conv1(x)[p] = b1 + sum_{o in 3x3} T[o][color(p+o)]   (gather-add, no GEMM)
//  2) 1x1 conv (linear) commutes with per-color mean pooling:
//     pooled[c] = (mean_{p:color=c} relu(conv1)) @ W2 + gate[c]*b2
//     -> per image only a [10,128]@[128,64] GEMM remains; done with
//        v_wmma_f32_16x16x32_f16 (padded to 16 rows, f16 in / f32 accumulate).
//  3) slot attention (5x64 matrices, 3 iters) is tiny -> f32 VALU in-LDS.
// ---------------------------------------------------------------------------

typedef _Float16 v16h __attribute__((ext_vector_type(16)));
typedef float    v8f  __attribute__((ext_vector_type(8)));

#define NBLK_THREADS 256
#define EPS_LN 1e-5f

__device__ __forceinline__ float sigmoidf_(float x) { return 1.0f / (1.0f + __expf(-x)); }

// ---------------------------------------------------------------------------
// Kernel 1: build color-conv table T[9][11][128] (f32) and f16 copy of W2.
// ---------------------------------------------------------------------------
__global__ void k_table(const float* __restrict__ enc_emb,     // [10,128]
                        const float* __restrict__ conv1_w,     // [3,3,128,128] HWIO
                        const float* __restrict__ conv2_w,     // [128,64]
                        float* __restrict__ Tg,                // [9,11,128]
                        _Float16* __restrict__ w2h)            // [128,64]
{
    const int tid = threadIdx.x;
    // T[o][c][d] = sum_ci emb[c][ci] * W1[o][ci][d]; c==10 -> 0 (zero padding)
    for (int idx = tid; idx < 9 * 11 * 128; idx += NBLK_THREADS) {
        const int o = idx / (11 * 128);
        const int c = (idx / 128) % 11;
        const int d = idx & 127;
        float acc = 0.0f;
        if (c < 10) {
            const float* e = enc_emb + c * 128;
            const float* w = conv1_w + o * 128 * 128 + d;  // W1[o][ci][d], stride 128 over ci
            for (int ci = 0; ci < 128; ++ci) acc += e[ci] * w[ci * 128];
        }
        Tg[idx] = acc;
    }
    for (int idx = tid; idx < 128 * 64; idx += NBLK_THREADS)
        w2h[idx] = (_Float16)conv2_w[idx];
}

// ---------------------------------------------------------------------------
// Kernel 2: one workgroup per image. Full pipeline.
// ---------------------------------------------------------------------------
struct PhaseA {                       // encoder phase
    float table[9 * 11 * 128];        // 50688 B
    int   pg[32 * 32];                // padded color grid (10 = zero pad)
};
struct PhaseB {                       // slot-attention phase
    float    cf[10 * 67];             // color features
    _Float16 ah[16 * 128];            // f16 A-tile (hbar rows, padded to 16)
    float    lncf[10 * 67];
    float    kf[10 * 64];
    float    vf[10 * 64];
    float    slots[5 * 64];
    float    prev[5 * 64];
    float    lns[5 * 64];
    float    q[5 * 64];               // reused as "updates"
    float    attn[5 * 10];
    float    gi[5 * 192];
    float    gh[5 * 192];
    float    m1[5 * 128];
    float    ps[64];
    float    h1[64];
    float    emb[128];
};
union PhaseU { PhaseA a; PhaseB b; };

__global__ void __launch_bounds__(NBLK_THREADS)
k_main(const int*   __restrict__ grids,        // [1024,30,30]
       const float* __restrict__ conv1_b,      // [128]
       const float* __restrict__ conv2_b,      // [64]
       const float* __restrict__ ln_in_s, const float* __restrict__ ln_in_b,   // [67]
       const float* __restrict__ Wk, const float* __restrict__ Wv,             // [67,64]
       const float* __restrict__ slots_init,   // [1024,5,64]
       const float* __restrict__ ln_slot_s, const float* __restrict__ ln_slot_b,
       const float* __restrict__ Wq,           // [64,64]
       const float* __restrict__ gru_wih, const float* __restrict__ gru_whh,   // [64,192]
       const float* __restrict__ gru_bih, const float* __restrict__ gru_bhh,   // [192]
       const float* __restrict__ ln_mlp_s, const float* __restrict__ ln_mlp_b,
       const float* __restrict__ mlp_w1, const float* __restrict__ mlp_b1,     // [64,128],[128]
       const float* __restrict__ mlp_w2, const float* __restrict__ mlp_b2,     // [128,64],[64]
       const float* __restrict__ proj_w1, const float* __restrict__ proj_b1,   // [64,64],[64]
       const float* __restrict__ proj_w2, const float* __restrict__ proj_b2,   // [64,128],[128]
       const float* __restrict__ Tg,           // [9,11,128] table
       const _Float16* __restrict__ w2h,       // [128,64] f16
       float* __restrict__ out)                // emb[1024*128] ++ slots[1024*320]
{
    const int b   = blockIdx.x;
    const int tid = threadIdx.x;
    const int lane = tid & 31;
    const int wid  = tid >> 5;

    __shared__ PhaseU u;
    __shared__ float s_hsum[10 * 128];
    __shared__ float s_cnt[10], s_cx[10], s_cy[10], s_inv[10], s_gate[10];

    // ---------------- Phase 0: stage LDS ----------------
    for (int idx = tid; idx < 1024; idx += NBLK_THREADS) {
        const int yy = idx >> 5, xx = idx & 31;
        int c = 10;
        if (yy >= 1 && yy <= 30 && xx >= 1 && xx <= 30)
            c = grids[b * 900 + (yy - 1) * 30 + (xx - 1)];
        u.a.pg[idx] = c;
    }
    for (int idx = tid; idx < 9 * 11 * 128; idx += NBLK_THREADS) u.a.table[idx] = Tg[idx];
    for (int idx = tid; idx < 1280; idx += NBLK_THREADS) s_hsum[idx] = 0.0f;
    if (tid < 10) { s_cnt[tid] = 0.f; s_cx[tid] = 0.f; s_cy[tid] = 0.f; }
    __syncthreads();

    // ---------------- Phase 1: table-conv + per-color accumulation ---------
    // each wave owns one pixel at a time; lane owns channel slice [lane*4, lane*4+4)
    const float4 bias4 = *(const float4*)(conv1_b + lane * 4);
    for (int p = wid; p < 900; p += 8) {
        const int y = p / 30, x = p % 30;
        float4 acc = bias4;
#pragma unroll
        for (int ky = 0; ky < 3; ++ky)
#pragma unroll
            for (int kx = 0; kx < 3; ++kx) {
                const int c = u.a.pg[(y + ky) * 32 + (x + kx)];   // wave-uniform -> LDS broadcast
                const float4 t = *(const float4*)(&u.a.table[((ky * 3 + kx) * 11 + c) * 128 + lane * 4]);
                acc.x += t.x; acc.y += t.y; acc.z += t.z; acc.w += t.w;
            }
        acc.x = fmaxf(acc.x, 0.f); acc.y = fmaxf(acc.y, 0.f);
        acc.z = fmaxf(acc.z, 0.f); acc.w = fmaxf(acc.w, 0.f);
        const int cc = u.a.pg[(y + 1) * 32 + (x + 1)];
        float* hs = &s_hsum[cc * 128 + lane * 4];
        atomicAdd(hs + 0, acc.x); atomicAdd(hs + 1, acc.y);
        atomicAdd(hs + 2, acc.z); atomicAdd(hs + 3, acc.w);
    }
    // counts & centroid sums
    for (int p = tid; p < 900; p += NBLK_THREADS) {
        const int c = u.a.pg[(p / 30 + 1) * 32 + (p % 30 + 1)];
        atomicAdd(&s_cnt[c], 1.0f);
        atomicAdd(&s_cx[c], (float)(p % 30) * (1.0f / 29.0f));
        atomicAdd(&s_cy[c], (float)(p / 30) * (1.0f / 29.0f));
    }
    __syncthreads();

    // ---------------- Phase 2: color features (WMMA pooled projection) -----
    if (tid < 10) {
        const float c  = s_cnt[tid];
        const float ms = fmaxf(c, 1e-8f);
        s_inv[tid]  = 1.0f / ms;
        s_gate[tid] = c / ms;                       // 1 if color present, else 0
        u.b.cf[tid * 67 + 64] = c * (1.0f / 900.0f);
        u.b.cf[tid * 67 + 65] = s_cx[tid] / ms;
        u.b.cf[tid * 67 + 66] = s_cy[tid] / ms;
    }
    __syncthreads();
    // hbar rows (f16), padded to 16 rows for the WMMA A-tile
    for (int idx = tid; idx < 16 * 128; idx += NBLK_THREADS) {
        const int m = idx >> 7, d = idx & 127;
        const float v = (m < 10) ? s_hsum[m * 128 + d] * s_inv[m] : 0.0f;
        u.b.ah[idx] = (_Float16)v;
    }
    __syncthreads();
    if (tid < 32) {                                  // wave 0, EXEC all-1s
        const int half = lane >> 4;
        const int col  = lane & 15;
#pragma unroll
        for (int nt = 0; nt < 4; ++nt) {             // N tiles of 16 (N=64)
            v8f acc = {0.f, 0.f, 0.f, 0.f, 0.f, 0.f, 0.f, 0.f};
#pragma unroll
            for (int kb = 0; kb < 4; ++kb) {         // K steps of 32 (K=128)
                v16h af, bf;
                // A (16x32 f16): VGPR i, elem j -> K = 16*(i/4) + 8*half + 2*(i%4) + j
#pragma unroll
                for (int i = 0; i < 8; ++i)
#pragma unroll
                    for (int j = 0; j < 2; ++j) {
                        const int K = kb * 32 + 16 * (i >> 2) + 8 * half + 2 * (i & 3) + j;
                        af[2 * i + j] = u.b.ah[col * 128 + K];
                    }
                // B (32x16 f16): elem e -> K = 16*(e/8) + 8*half + (e%8) (same as A); col = lane&15
#pragma unroll
                for (int e = 0; e < 16; ++e) {
                    const int K = kb * 32 + 16 * (e >> 3) + 8 * half + (e & 7);
                    bf[e] = w2h[K * 64 + nt * 16 + col];
                }
                acc = __builtin_amdgcn_wmma_f32_16x16x32_f16(
                    false, af, false, bf, (short)0, acc, false, false);
            }
            // D (16x16 f32): VGPR r -> row = r + 8*half, col = lane&15
#pragma unroll
            for (int r = 0; r < 8; ++r) {
                const int m = r + 8 * half;
                if (m < 10) {
                    const int n = nt * 16 + col;
                    u.b.cf[m * 67 + n] = acc[r] + s_gate[m] * conv2_b[n];
                }
            }
        }
    }
    __syncthreads();

    // ---------------- Phase 3: slot attention (f32 VALU, in-LDS) ----------
    // LN over color features (67)
    if (tid < 10) {
        const float* row = &u.b.cf[tid * 67];
        float m = 0.f; for (int e = 0; e < 67; ++e) m += row[e];
        m *= (1.0f / 67.0f);
        float v = 0.f; for (int e = 0; e < 67; ++e) { const float d = row[e] - m; v += d * d; }
        v *= (1.0f / 67.0f);
        const float rs = rsqrtf(v + EPS_LN);
        for (int e = 0; e < 67; ++e)
            u.b.lncf[tid * 67 + e] = (row[e] - m) * rs * ln_in_s[e] + ln_in_b[e];
    }
    __syncthreads();
    for (int idx = tid; idx < 640; idx += NBLK_THREADS) {   // k, v projections
        const int c = idx >> 6, d = idx & 63;
        const float* x = &u.b.lncf[c * 67];
        float ak = 0.f, av = 0.f;
        for (int e = 0; e < 67; ++e) { ak += x[e] * Wk[e * 64 + d]; av += x[e] * Wv[e * 64 + d]; }
        u.b.kf[idx] = ak; u.b.vf[idx] = av;
    }
    for (int idx = tid; idx < 320; idx += NBLK_THREADS)
        u.b.slots[idx] = slots_init[b * 320 + idx];
    __syncthreads();

    for (int it = 0; it < 3; ++it) {
        for (int idx = tid; idx < 320; idx += NBLK_THREADS) u.b.prev[idx] = u.b.slots[idx];
        __syncthreads();
        if (tid < 5) {                                   // LN(slots)
            const float* row = &u.b.slots[tid * 64];
            float m = 0.f; for (int e = 0; e < 64; ++e) m += row[e];
            m *= (1.0f / 64.0f);
            float v = 0.f; for (int e = 0; e < 64; ++e) { const float d = row[e] - m; v += d * d; }
            v *= (1.0f / 64.0f);
            const float rs = rsqrtf(v + EPS_LN);
            for (int e = 0; e < 64; ++e)
                u.b.lns[tid * 64 + e] = (row[e] - m) * rs * ln_slot_s[e] + ln_slot_b[e];
        }
        __syncthreads();
        for (int idx = tid; idx < 320; idx += NBLK_THREADS) {   // q = LN(slots) @ Wq
            const int s = idx >> 6, d = idx & 63;
            float a = 0.f;
            for (int e = 0; e < 64; ++e) a += u.b.lns[s * 64 + e] * Wq[e * 64 + d];
            u.b.q[idx] = a;
        }
        __syncthreads();
        for (int idx = tid; idx < 50; idx += NBLK_THREADS) {    // logits = q.k * D^-1/2
            const int s = idx / 10, c = idx % 10;
            float a = 0.f;
            for (int e = 0; e < 64; ++e) a += u.b.q[s * 64 + e] * u.b.kf[c * 64 + e];
            u.b.attn[idx] = a * 0.125f;
        }
        __syncthreads();
        if (tid < 10) {                                  // softmax over slots (per color)
            const int c = tid;
            float mx = -1e30f;
            for (int s = 0; s < 5; ++s) mx = fmaxf(mx, u.b.attn[s * 10 + c]);
            float sm = 0.f;
            for (int s = 0; s < 5; ++s) { const float e = __expf(u.b.attn[s * 10 + c] - mx); u.b.attn[s * 10 + c] = e; sm += e; }
            const float inv = 1.0f / sm;
            for (int s = 0; s < 5; ++s) u.b.attn[s * 10 + c] *= inv;
        }
        __syncthreads();
        if (tid < 5) {                                   // weighted mean over inputs
            float sm = 1e-8f;
            for (int c = 0; c < 10; ++c) sm += u.b.attn[tid * 10 + c];
            const float inv = 1.0f / sm;
            for (int c = 0; c < 10; ++c) u.b.attn[tid * 10 + c] *= inv;
        }
        __syncthreads();
        for (int idx = tid; idx < 320; idx += NBLK_THREADS) {   // updates = attn @ v  (reuse q)
            const int s = idx >> 6, d = idx & 63;
            float a = 0.f;
            for (int c = 0; c < 10; ++c) a += u.b.attn[s * 10 + c] * u.b.vf[c * 64 + d];
            u.b.q[idx] = a;
        }
        __syncthreads();
        for (int idx = tid; idx < 960; idx += NBLK_THREADS) {   // GRU gates
            const int s = idx / 192, j = idx % 192;
            float a = gru_bih[j], h = gru_bhh[j];
            for (int e = 0; e < 64; ++e) {
                a += u.b.q[s * 64 + e] * gru_wih[e * 192 + j];
                h += u.b.prev[s * 64 + e] * gru_whh[e * 192 + j];
            }
            u.b.gi[idx] = a; u.b.gh[idx] = h;
        }
        __syncthreads();
        for (int idx = tid; idx < 320; idx += NBLK_THREADS) {   // GRU combine
            const int s = idx >> 6, d = idx & 63;
            const float ir = u.b.gi[s * 192 + d],        hr = u.b.gh[s * 192 + d];
            const float iz = u.b.gi[s * 192 + 64 + d],   hz = u.b.gh[s * 192 + 64 + d];
            const float in = u.b.gi[s * 192 + 128 + d],  hn = u.b.gh[s * 192 + 128 + d];
            const float r = sigmoidf_(ir + hr);
            const float z = sigmoidf_(iz + hz);
            const float n = tanhf(in + r * hn);
            u.b.slots[idx] = (1.0f - z) * n + z * u.b.prev[idx];
        }
        __syncthreads();
        if (tid < 5) {                                   // LN for MLP
            const float* row = &u.b.slots[tid * 64];
            float m = 0.f; for (int e = 0; e < 64; ++e) m += row[e];
            m *= (1.0f / 64.0f);
            float v = 0.f; for (int e = 0; e < 64; ++e) { const float d = row[e] - m; v += d * d; }
            v *= (1.0f / 64.0f);
            const float rs = rsqrtf(v + EPS_LN);
            for (int e = 0; e < 64; ++e)
                u.b.lns[tid * 64 + e] = (row[e] - m) * rs * ln_mlp_s[e] + ln_mlp_b[e];
        }
        __syncthreads();
        for (int idx = tid; idx < 640; idx += NBLK_THREADS) {   // MLP layer 1 + ReLU
            const int s = idx >> 7, j = idx & 127;
            float a = mlp_b1[j];
            for (int e = 0; e < 64; ++e) a += u.b.lns[s * 64 + e] * mlp_w1[e * 128 + j];
            u.b.m1[idx] = fmaxf(a, 0.f);
        }
        __syncthreads();
        for (int idx = tid; idx < 320; idx += NBLK_THREADS) {   // MLP layer 2, residual
            const int s = idx >> 6, d = idx & 63;
            float a = mlp_b2[d];
            for (int j = 0; j < 128; ++j) a += u.b.m1[s * 128 + j] * mlp_w2[j * 64 + d];
            u.b.slots[idx] += a;
        }
        __syncthreads();
    }

    // ---------------- Phase 4: outputs -------------------------------------
    for (int idx = tid; idx < 320; idx += NBLK_THREADS)
        out[1024 * 128 + b * 320 + idx] = u.b.slots[idx];
    if (tid < 64) {                                      // pooled slots
        float a = 0.f;
        for (int s = 0; s < 5; ++s) a += u.b.slots[s * 64 + tid];
        u.b.ps[tid] = a * 0.2f;
    }
    __syncthreads();
    if (tid < 64) {                                      // proj layer 1 + ReLU
        float a = proj_b1[tid];
        for (int e = 0; e < 64; ++e) a += u.b.ps[e] * proj_w1[e * 64 + tid];
        u.b.h1[tid] = fmaxf(a, 0.f);
    }
    __syncthreads();
    for (int idx = tid; idx < 128; idx += NBLK_THREADS) {  // proj layer 2
        float a = proj_b2[idx];
        for (int e = 0; e < 64; ++e) a += u.b.h1[e] * proj_w2[e * 128 + idx];
        u.b.emb[idx] = a;
    }
    __syncthreads();
    if (tid < 128) {                                     // L2 normalize + write
        float ss = 0.f;
        for (int j = 0; j < 128; ++j) ss += u.b.emb[j] * u.b.emb[j];
        out[b * 128 + tid] = u.b.emb[tid] / fmaxf(sqrtf(ss), 1e-12f);
    }
}

// ---------------------------------------------------------------------------
extern "C" void kernel_launch(void* const* d_in, const int* in_sizes, int n_in,
                              void* d_out, int out_size, void* d_ws, size_t ws_size,
                              hipStream_t stream) {
    (void)in_sizes; (void)n_in; (void)out_size; (void)ws_size;
    const int*   grids       = (const int*)  d_in[0];
    const float* enc_emb     = (const float*)d_in[1];
    const float* enc_conv1_w = (const float*)d_in[2];
    const float* enc_conv1_b = (const float*)d_in[3];
    const float* enc_conv2_w = (const float*)d_in[4];
    const float* enc_conv2_b = (const float*)d_in[5];
    const float* ln_in_s     = (const float*)d_in[6];
    const float* ln_in_b     = (const float*)d_in[7];
    const float* Wk          = (const float*)d_in[8];
    const float* Wv          = (const float*)d_in[9];
    const float* slots_init  = (const float*)d_in[10];
    const float* ln_slot_s   = (const float*)d_in[11];
    const float* ln_slot_b   = (const float*)d_in[12];
    const float* Wq          = (const float*)d_in[13];
    const float* gru_wih     = (const float*)d_in[14];
    const float* gru_whh     = (const float*)d_in[15];
    const float* gru_bih     = (const float*)d_in[16];
    const float* gru_bhh     = (const float*)d_in[17];
    const float* ln_mlp_s    = (const float*)d_in[18];
    const float* ln_mlp_b    = (const float*)d_in[19];
    const float* mlp_w1      = (const float*)d_in[20];
    const float* mlp_b1      = (const float*)d_in[21];
    const float* mlp_w2      = (const float*)d_in[22];
    const float* mlp_b2      = (const float*)d_in[23];
    const float* proj_w1     = (const float*)d_in[24];
    const float* proj_b1     = (const float*)d_in[25];
    const float* proj_w2     = (const float*)d_in[26];
    const float* proj_b2     = (const float*)d_in[27];

    float*    Tg  = (float*)d_ws;                                  // 9*11*128 f32
    _Float16* w2h = (_Float16*)((char*)d_ws + 9 * 11 * 128 * 4);   // 128*64 f16

    k_table<<<1, NBLK_THREADS, 0, stream>>>(enc_emb, enc_conv1_w, enc_conv2_w, Tg, w2h);
    k_main<<<1024, NBLK_THREADS, 0, stream>>>(
        grids, enc_conv1_b, enc_conv2_b, ln_in_s, ln_in_b, Wk, Wv, slots_init,
        ln_slot_s, ln_slot_b, Wq, gru_wih, gru_whh, gru_bih, gru_bhh,
        ln_mlp_s, ln_mlp_b, mlp_w1, mlp_b1, mlp_w2, mlp_b2,
        proj_w1, proj_b1, proj_w2, proj_b2, Tg, w2h, (float*)d_out);
}
